// SAGE_42331197669872
// MI455X (gfx1250) — matrix operand order repaired and checked by reading, not compiled, not run on hardware
//
#include <hip/hip_runtime.h>
#include <math.h>

#define NNODES 100000
#define NEDGES 1600000
#define NCLASS 40
#define KDIM   128
#define KP     136   // padded LDS row stride (bf16 elems): 272B, 16B-aligned, bank-rotating

typedef __bf16 bf16_t;
typedef __attribute__((ext_vector_type(16))) __bf16 v16bf;
typedef __attribute__((ext_vector_type(8)))  __bf16 v8bf;
typedef __attribute__((ext_vector_type(4)))  __bf16 v4bf;
typedef __attribute__((ext_vector_type(8)))  float  v8f;

// ---------------------------------------------------------------------------
// GEMM: H[N,F] = X[N,128] @ W[128,F] + bias   (bf16 WMMA, f32 accumulate)
// Block = 256 threads (8 waves), covers 32 rows x Fpad cols.
// A-frag: two contiguous 16B runs  -> ds_load_b128 x2
// B-frag: W staged transposed in LDS -> two 16B runs -> ds_load_b128 x2
// K loop fully unrolled -> 4 x v_wmma_f32_16x16x32_bf16 per 16x16 tile.
// ---------------------------------------------------------------------------
__global__ void __launch_bounds__(256)
gemm_bias_wmma(const float* __restrict__ X, const float* __restrict__ W,
               const float* __restrict__ bias, float* __restrict__ H,
               int N, int F, int Fpad) {
  __shared__ bf16_t sX[32 * KP];     // [row][k]   (padded stride)
  __shared__ bf16_t sWt[128 * KP];   // [col][k]   transposed, padded stride
  const int tid = threadIdx.x;
  const int rowBase = blockIdx.x * 32;

  // Stage X tile: 32 rows x 128 cols, float4-coalesced, packed bf16x4 stores.
  for (int i4 = tid; i4 < 32 * 32; i4 += 256) {     // 32 float4 per row
    int r = i4 >> 5, c4 = i4 & 31;
    int gr = rowBase + r;
    float4 v = make_float4(0.f, 0.f, 0.f, 0.f);
    if (gr < N) v = ((const float4*)(X + (size_t)gr * KDIM))[c4];
    v4bf t = {(bf16_t)v.x, (bf16_t)v.y, (bf16_t)v.z, (bf16_t)v.w};
    *(v4bf*)&sX[r * KP + c4 * 4] = t;
  }
  // Stage W transposed: sWt[f][k] = W[k][f], coalesced on f, zero-pad f>=F.
  for (int i = tid; i < KDIM * Fpad; i += 256) {
    int k = i / Fpad, f = i - k * Fpad;
    float v = (f < F) ? W[(size_t)k * F + f] : 0.0f;
    sWt[f * KP + k] = (bf16_t)v;
  }
  __syncthreads();

  const int wave = tid >> 5;
  const int lane = tid & 31;
  const int hi   = (lane >> 4) & 1;
  const int ln   = lane & 15;          // M for A frag, N (col) for B frag
  const int nColT  = Fpad >> 4;
  const int nTiles = 2 * nColT;

  for (int t = wave; t < nTiles; t += 8) {
    const int rt = t / nColT;
    const int ct = t - rt * nColT;
    v8f acc = {};
    const bf16_t* ax = &sX[(rt * 16 + ln) * KP];
    const bf16_t* bx = &sWt[(ct * 16 + ln) * KP];
#pragma unroll
    for (int kk = 0; kk < 4; ++kk) {
      const int k0 = kk * 32;
      // A: j<8 -> K=k0+j+8hi ; j>=8 -> K=k0+16+(j-8)+8hi
      v8bf alo = *(const v8bf*)(ax + k0 + 8 * hi);
      v8bf ahi = *(const v8bf*)(ax + k0 + 16 + 8 * hi);
      v16bf a = __builtin_shufflevector(alo, ahi, 0, 1, 2, 3, 4, 5, 6, 7,
                                        8, 9, 10, 11, 12, 13, 14, 15);
      // B: j<8 -> K=k0+j+8hi ; j>=8 -> K=k0+16+(j-8)+8hi
      v8bf blo = *(const v8bf*)(bx + k0 + 8 * hi);
      v8bf bhi = *(const v8bf*)(bx + k0 + 16 + 8 * hi);
      v16bf b = __builtin_shufflevector(blo, bhi, 0, 1, 2, 3, 4, 5, 6, 7,
                                        8, 9, 10, 11, 12, 13, 14, 15);
      acc = __builtin_amdgcn_wmma_f32_16x16x32_bf16(
          false, a, false, b, (short)0, acc, false, false);
    }
    const int col = ct * 16 + ln;
    const float bv = (col < F) ? bias[col] : 0.0f;
    const int row0 = rowBase + rt * 16 + 8 * hi;     // C/D: VGPR r -> M=r(+8)
    float* hp = H + (size_t)row0 * F + col;
#pragma unroll
    for (int r = 0; r < 8; ++r) {
      if (row0 + r < N && col < F)
        hp[(size_t)r * F] = acc[r] + bv;
    }
  }
}

// ---------------------------------------------------------------------------
// Edge scatter (F==128 fast path): one wave per edge, float4 per lane.
// agg accumulator (51 MB) lives in L2, so raw global f32 atomics are cheap.
// ---------------------------------------------------------------------------
__global__ void __launch_bounds__(256)
edge_scatter_f128(const float* __restrict__ H, const int* __restrict__ src,
                  const int* __restrict__ dst, float* __restrict__ agg,
                  float* __restrict__ cnt, int E) {
  int e = blockIdx.x * 8 + (threadIdx.x >> 5);
  if (e >= E) return;
  int lane = threadIdx.x & 31;
  int s = src[e], d = dst[e];
  if ((unsigned)s >= (unsigned)NNODES || (unsigned)d >= (unsigned)NNODES) return;
  const float4 v = ((const float4*)(H + (size_t)s * 128))[lane];
  float* ad = agg + (size_t)d * 128 + lane * 4;
  atomicAdd(ad + 0, v.x);
  atomicAdd(ad + 1, v.y);
  atomicAdd(ad + 2, v.z);
  atomicAdd(ad + 3, v.w);
  if (lane == 0) atomicAdd(cnt + d, 1.0f);
}

__global__ void __launch_bounds__(256)
edge_scatter_gen(const float* __restrict__ H, const int* __restrict__ src,
                 const int* __restrict__ dst, float* __restrict__ agg,
                 float* __restrict__ cnt, int E, int F) {
  int e = blockIdx.x * 8 + (threadIdx.x >> 5);
  if (e >= E) return;
  int lane = threadIdx.x & 31;
  int s = src[e], d = dst[e];
  if ((unsigned)s >= (unsigned)NNODES || (unsigned)d >= (unsigned)NNODES) return;
  for (int f = lane; f < F; f += 32)
    atomicAdd(agg + (size_t)d * F + f, H[(size_t)s * F + f]);
  if (lane == 0) atomicAdd(cnt + d, 1.0f);
}

// ---------------------------------------------------------------------------
// Label scatter: LDS-privatized partial sums (ds_add_f32), one flush per block.
// ---------------------------------------------------------------------------
__global__ void __launch_bounds__(256)
label_scatter(const float* __restrict__ H, const int* __restrict__ label,
              float* __restrict__ csum, float* __restrict__ ccnt,
              int N, int F, int nodesPerBlock) {
  __shared__ float ls[NCLASS * 128];
  __shared__ float lc[NCLASS];
  const int tid = threadIdx.x;
  for (int i = tid; i < NCLASS * F; i += 256) ls[i] = 0.0f;
  if (tid < NCLASS) lc[tid] = 0.0f;
  __syncthreads();

  const int start = blockIdx.x * nodesPerBlock;
  const int end   = min(start + nodesPerBlock, N);
  const int wave = tid >> 5, lane = tid & 31;
  for (int n = start + wave; n < end; n += 8) {
    int l = label[n];
    if ((unsigned)l >= (unsigned)NCLASS) continue;
    for (int f = lane; f < F; f += 32)
      atomicAdd(&ls[l * F + f], H[(size_t)n * F + f]);
    if (lane == 0) atomicAdd(&lc[l], 1.0f);
  }
  __syncthreads();
  for (int i = tid; i < NCLASS * F; i += 256) atomicAdd(&csum[i], ls[i]);
  if (tid < NCLASS) atomicAdd(&ccnt[tid], lc[tid]);
}

// ---------------------------------------------------------------------------
// centers[c][f] = csum[c][f] / max(ccnt[c], 1)
// ---------------------------------------------------------------------------
__global__ void centers_kernel(const float* __restrict__ csum,
                               const float* __restrict__ ccnt,
                               float* __restrict__ cent, int F) {
  int i = blockIdx.x * blockDim.x + threadIdx.x;
  if (i < NCLASS * F) {
    int c = i / F;
    cent[i] = csum[i] / fmaxf(ccnt[c], 1.0f);
  }
}

// ---------------------------------------------------------------------------
// rmat[a][f] = sum_b cm[a,0,b] * normalize(cent[a]-cent[b])[f]
// One block per class 'a'; 128 threads; LDS tree reduction for the norm.
// ---------------------------------------------------------------------------
__global__ void __launch_bounds__(128)
rmat_kernel(const float* __restrict__ cent, const float* __restrict__ cm,
            float* __restrict__ rmat, int F) {
  __shared__ float red[128];
  const int a = blockIdx.x;
  const int t = threadIdx.x;
  float ca  = (t < F) ? cent[a * F + t] : 0.0f;
  float acc = 0.0f;
  for (int b = 0; b < NCLASS; ++b) {
    float diff = (t < F) ? (ca - cent[b * F + t]) : 0.0f;
    red[t] = diff * diff;
    __syncthreads();
    for (int s = 64; s > 0; s >>= 1) {
      if (t < s) red[t] += red[t + s];
      __syncthreads();
    }
    float rn = sqrtf(red[0]);
    float scale = cm[a * NCLASS + b] / ((rn == 0.0f) ? 1.0f : rn);
    acc = fmaf(scale, diff, acc);
    __syncthreads();
  }
  if (t < F) rmat[a * F + t] = acc;
}

// ---------------------------------------------------------------------------
// out[n][f] = h + 0.5*(p[n]@rmat)[f] + 0.5*(agg[n][f]/max(cnt,1) - h); relu opt
// rmat cached in LDS; p-row cached in registers per wave (40 FMAs/output).
// ---------------------------------------------------------------------------
__global__ void __launch_bounds__(256)
combine_kernel(const float* __restrict__ H, const float* __restrict__ AGG,
               const float* __restrict__ CNT, const float* __restrict__ P,
               const float* __restrict__ RMAT, float* __restrict__ out,
               int N, int F, int relu) {
  __shared__ float sr[NCLASS * 128];
  const int tid = threadIdx.x;
  for (int i = tid; i < NCLASS * F; i += 256) sr[i] = RMAT[i];
  __syncthreads();

  const int lane   = tid & 31;
  const int wave   = blockIdx.x * 8 + (tid >> 5);
  const int nwaves = gridDim.x * 8;
  for (int n = wave; n < N; n += nwaves) {
    float inv = 1.0f / fmaxf(CNT[n], 1.0f);
    const float* pn = P + (size_t)n * NCLASS;
    float pr[NCLASS];
#pragma unroll
    for (int c = 0; c < NCLASS; ++c) pr[c] = pn[c];
    for (int f = lane; f < F; f += 32) {
      float r = 0.0f;
#pragma unroll
      for (int c = 0; c < NCLASS; ++c) r = fmaf(pr[c], sr[c * F + f], r);
      float hv = H[(size_t)n * F + f];
      float v  = hv + 0.5f * r + 0.5f * (AGG[(size_t)n * F + f] * inv - hv);
      out[(size_t)n * F + f] = relu ? fmaxf(v, 0.0f) : v;
    }
  }
}

// ---------------------------------------------------------------------------
// Host-side layer driver (all launches on `stream`; memsets are capturable)
// ---------------------------------------------------------------------------
static void run_layer(const float* xin, const float* W, const float* bias,
                      const int* src, const int* dst, const int* label,
                      const float* p, const float* cm,
                      float* hbuf, float* outbuf,
                      float* AGG, float* CNT, float* CSUM, float* CCNT,
                      float* CENT, float* RMAT,
                      int F, int relu, hipStream_t stream) {
  const int Fpad = ((F + 15) / 16) * 16;
  gemm_bias_wmma<<<(NNODES + 31) / 32, 256, 0, stream>>>(
      xin, W, bias, hbuf, NNODES, F, Fpad);

  hipMemsetAsync(AGG,  0, (size_t)NNODES * F * sizeof(float), stream);
  hipMemsetAsync(CNT,  0, (size_t)NNODES * sizeof(float), stream);
  hipMemsetAsync(CSUM, 0, (size_t)NCLASS * F * sizeof(float), stream);
  hipMemsetAsync(CCNT, 0, (size_t)NCLASS * sizeof(float), stream);

  if (F == 128)
    edge_scatter_f128<<<(NEDGES + 7) / 8, 256, 0, stream>>>(
        hbuf, src, dst, AGG, CNT, NEDGES);
  else
    edge_scatter_gen<<<(NEDGES + 7) / 8, 256, 0, stream>>>(
        hbuf, src, dst, AGG, CNT, NEDGES, F);

  const int nodesPerBlock = 512;
  label_scatter<<<(NNODES + nodesPerBlock - 1) / nodesPerBlock, 256, 0, stream>>>(
      hbuf, label, CSUM, CCNT, NNODES, F, nodesPerBlock);

  centers_kernel<<<(NCLASS * F + 255) / 256, 256, 0, stream>>>(CSUM, CCNT, CENT, F);
  rmat_kernel<<<NCLASS, 128, 0, stream>>>(CENT, cm, RMAT, F);

  combine_kernel<<<1024, 256, 0, stream>>>(
      hbuf, AGG, CNT, p, RMAT, outbuf, NNODES, F, relu);
}

extern "C" void kernel_launch(void* const* d_in, const int* in_sizes, int n_in,
                              void* d_out, int out_size, void* d_ws, size_t ws_size,
                              hipStream_t stream) {
  const float* x     = (const float*)d_in[0];
  const int*   ei    = (const int*)d_in[1];
  const int*   label = (const int*)d_in[2];
  const float* p     = (const float*)d_in[3];
  const float* cm    = (const float*)d_in[4];
  const float* W1    = (const float*)d_in[5];
  const float* b1    = (const float*)d_in[6];
  const float* W2    = (const float*)d_in[7];
  const float* b2    = (const float*)d_in[8];
  const float* W3    = (const float*)d_in[9];
  const float* b3    = (const float*)d_in[10];
  const int* src = ei;
  const int* dst = ei + NEDGES;

  float* wsf  = (float*)d_ws;
  float* A    = wsf;                                // N*128  feature buf (ping)
  float* Bb   = A   + (size_t)NNODES * 128;         // N*128  feature buf (pong)
  float* AGG  = Bb  + (size_t)NNODES * 128;         // N*128  scatter accumulator
  float* CNT  = AGG + (size_t)NNODES * 128;         // N      degree counts
  float* CSUM = CNT + NNODES;                       // 40*128 class sums
  float* CCNT = CSUM + NCLASS * 128;                // 40     class counts
  float* CENT = CCNT + 64;                          // 40*128 class centers
  float* RMAT = CENT + NCLASS * 128;                // 40*128 rectify matrix

  // Layer 1: 128 -> 128, relu; in-place combine into A
  run_layer(x,  W1, b1, src, dst, label, p, cm,
            A,  A,  AGG, CNT, CSUM, CCNT, CENT, RMAT, 128, 1, stream);
  // Layer 2: 128 -> 128, relu; combine into B
  run_layer(A,  W2, b2, src, dst, label, p, cm,
            Bb, Bb, AGG, CNT, CSUM, CCNT, CENT, RMAT, 128, 1, stream);
  // Layer 3: 128 -> 40, no relu; h staged in A (reuse), final to d_out
  run_layer(Bb, W3, b3, src, dst, label, p, cm,
            A,  (float*)d_out, AGG, CNT, CSUM, CCNT, CENT, RMAT, NCLASS, 0, stream);
}
